// GeometryAwareTensorAttention_73452530696702
// MI455X (gfx1250) — matrix operand order repaired and checked by reading, not compiled, not run on hardware
//
#include <hip/hip_runtime.h>

typedef __bf16 bf16;
typedef __attribute__((ext_vector_type(16))) __bf16 v16bf;
typedef __attribute__((ext_vector_type(8)))  float  v8f;

// ---------------- problem constants ----------------
static constexpr int N_    = 128;   // sequence
static constexpr int DIMc  = 64;    // DIM
static constexpr int DI    = 512;   // DIM_INNER
static constexpr int DMLP  = 1024;  // DIM_MLP
static constexpr int SDI   = 2560;  // S*DIM_INNER
static constexpr int S_    = 5;
static constexpr int H_    = 8;

// ---------------- workspace layout (bytes) ----------------
static constexpr size_t O_HI   = 0;
static constexpr size_t O_HJ   = O_HI  + (size_t)N_*DIMc*2;
static constexpr size_t O_T    = O_HJ  + (size_t)N_*DIMc*2;
static constexpr size_t O_WQ   = O_T   + (size_t)N_*N_*DIMc*2;      // t_ij bf16
static constexpr size_t O_WK   = O_WQ  + (size_t)DI*DIMc*2;          // W_q^T bf16 (512x64)
static constexpr size_t O_WV1  = O_WK  + (size_t)DI*DIMc*2;
static constexpr size_t O_WP1  = O_WV1 + (size_t)DMLP*DIMc*2;        // (1024x64)
static constexpr size_t O_WV2  = O_WP1 + (size_t)DMLP*DIMc*2;
static constexpr size_t O_WP2  = O_WV2 + (size_t)SDI*DMLP*2;         // (2560x1024)
static constexpr size_t O_WEK  = O_WP2 + (size_t)SDI*DMLP*2;
static constexpr size_t O_WEV  = O_WEK + (size_t)SDI*DIMc*2;         // (2560x64)
static constexpr size_t O_WO   = O_WEV + (size_t)SDI*DIMc*2;
static constexpr size_t O_Q    = O_WO  + (size_t)DIMc*DI*2;          // W_o^T bf16 (64x512)
static constexpr size_t O_K    = O_Q   + (size_t)N_*DI*4;            // q fp32 (128x512)
static constexpr size_t O_A1V  = O_K   + (size_t)N_*DI*4;
static constexpr size_t O_A1P  = O_A1V + (size_t)N_*DMLP*2;          // silu act bf16
static constexpr size_t O_V    = O_A1P + (size_t)N_*DMLP*2;
static constexpr size_t O_P    = O_V   + (size_t)N_*SDI*4;           // v fp32 (128x2560)
static constexpr size_t O_ATTN = O_P   + (size_t)N_*SDI*4;           // attn fp32 [i][s][h][j]

// ---------------- WMMA fragment helpers (wave32, 16x16x32 bf16) ----------------
// A (16x32, row-major src): lane L -> row L%16 ; elem e -> K = e + 8*(e>=8) + 8*(L>=16)
__device__ inline v16bf frag_a(const bf16* __restrict__ base, int ld, int k0) {
  int lane = threadIdx.x & 31;
  const bf16* r = base + (size_t)(lane & 15) * ld + k0 + ((lane >> 4) << 3);
  v16bf f;
#pragma unroll
  for (int e = 0; e < 8; ++e) { f[e] = r[e]; f[8 + e] = r[16 + e]; }
  return f;
}
// B (32x16) from N-major (transposed) src: lane L -> col L%16 ; elem e -> K = k0 + e + 8*(e>=8) + 8*(L>=16)
__device__ inline v16bf frag_b(const bf16* __restrict__ base, int ld, int k0) {
  int lane = threadIdx.x & 31;
  const bf16* r = base + (size_t)(lane & 15) * ld + k0 + ((lane >> 4) << 3);
  v16bf f;
#pragma unroll
  for (int e = 0; e < 8; ++e) { f[e] = r[e]; f[8 + e] = r[16 + e]; }
  return f;
}
__device__ inline v8f wmma_bf(v16bf a, v16bf b, v8f c) {
  return __builtin_amdgcn_wmma_f32_16x16x32_bf16(false, a, false, b, (short)0, c, false, false);
}
// D (16x16 f32): lane L, reg r -> row = r + 8*(L>=16), col = L%16
__device__ inline v8f gemm_tile(const bf16* __restrict__ A, int lda,
                                const bf16* __restrict__ B, int ldb, int K) {
  v8f c = {0.f,0.f,0.f,0.f,0.f,0.f,0.f,0.f};
  for (int k0 = 0; k0 < K; k0 += 32)
    c = wmma_bf(frag_a(A, lda, k0), frag_b(B, ldb, k0), c);
  return c;
}

__device__ inline float silu_f(float x) { return x / (1.f + __expf(-x)); }

// ---------------- prep kernels ----------------
__global__ void ln_kernel(const float* __restrict__ h, const float* __restrict__ gi,
                          const float* __restrict__ gj, bf16* hi, bf16* hj) {
  __shared__ float red[64];
  int row = blockIdx.x, t = threadIdx.x;
  float x = h[row * 64 + t];
  red[t] = x; __syncthreads();
  for (int s = 32; s > 0; s >>= 1) { if (t < s) red[t] += red[t + s]; __syncthreads(); }
  float mean = red[0] * (1.f / 64.f);
  __syncthreads();
  float d = x - mean;
  red[t] = d * d; __syncthreads();
  for (int s = 32; s > 0; s >>= 1) { if (t < s) red[t] += red[t + s]; __syncthreads(); }
  float var = red[0] * (1.f / 64.f);
  float y = d * rsqrtf(var + 1e-5f);
  hi[row * 64 + t] = (bf16)(y * gi[t]);
  hj[row * 64 + t] = (bf16)(y * gj[t]);
}

__global__ void cvt_bf16(const float* __restrict__ src, bf16* __restrict__ dst, int total) {
  int i = blockIdx.x * blockDim.x + threadIdx.x;
  if (i < total) dst[i] = (bf16)src[i];
}

// src KxN row-major (fp32) -> dst NxK row-major (bf16)
__global__ void transpose_bf16(const float* __restrict__ src, bf16* __restrict__ dst,
                               int K, int N) {
  int i = blockIdx.x * blockDim.x + threadIdx.x;
  if (i < K * N) {
    int n = i / K, k = i % K;
    dst[i] = (bf16)src[(size_t)k * N + n];
  }
}

// ---------------- small GEMMs: q/k, MLP ----------------
// out(128 x N) = act(128x64) @ Wt(Nx64)^T   [fp32 out]
__global__ __launch_bounds__(256) void qk_kernel(const bf16* __restrict__ act,
                                                 const bf16* __restrict__ wt,
                                                 float* __restrict__ out, int Ncols) {
  int w = threadIdx.x >> 5, lane = threadIdx.x & 31;
  int mt = blockIdx.x, nt = blockIdx.y * 8 + w;
  v8f d = gemm_tile(act + (size_t)mt * 16 * 64, 64, wt + (size_t)nt * 16 * 64, 64, 64);
  int n = nt * 16 + (lane & 15);
  int m0 = mt * 16 + ((lane >> 4) << 3);
#pragma unroll
  for (int r = 0; r < 8; ++r) out[(size_t)(m0 + r) * Ncols + n] = d[r];
}

// a1 = silu(act @ W1^T + b1)  bf16 out (128x1024)
__global__ __launch_bounds__(256) void mlp1_kernel(const bf16* __restrict__ act,
                                                   const bf16* __restrict__ wt,
                                                   const float* __restrict__ bias,
                                                   bf16* __restrict__ out) {
  int w = threadIdx.x >> 5, lane = threadIdx.x & 31;
  int mt = blockIdx.x, nt = blockIdx.y * 8 + w;
  v8f d = gemm_tile(act + (size_t)mt * 16 * 64, 64, wt + (size_t)nt * 16 * 64, 64, 64);
  int n = nt * 16 + (lane & 15);
  int m0 = mt * 16 + ((lane >> 4) << 3);
  float b = bias[n];
#pragma unroll
  for (int r = 0; r < 8; ++r)
    out[(size_t)(m0 + r) * DMLP + n] = (bf16)silu_f(d[r] + b);
}

// out = a1 @ W2^T + b2  fp32 (128x2560), K=1024; a1 row-tile staged in LDS
__global__ __launch_bounds__(256) void mlp2_kernel(const bf16* __restrict__ a1,
                                                   const bf16* __restrict__ wt,
                                                   const float* __restrict__ bias,
                                                   float* __restrict__ out) {
  __shared__ bf16 al[16 * DMLP];   // 32 KB
  int w = threadIdx.x >> 5, lane = threadIdx.x & 31;
  int mt = blockIdx.x, nt = blockIdx.y * 8 + w;
  {
    const unsigned* s = (const unsigned*)(a1 + (size_t)mt * 16 * DMLP);
    unsigned* du = (unsigned*)al;
    for (int t = threadIdx.x; t < 16 * DMLP / 2; t += 256) du[t] = s[t];
  }
  __syncthreads();
  v8f d = gemm_tile(al, DMLP, wt + (size_t)nt * 16 * DMLP, DMLP, DMLP);
  int n = nt * 16 + (lane & 15);
  int m0 = mt * 16 + ((lane >> 4) << 3);
  float b = bias[n];
#pragma unroll
  for (int r = 0; r < 8; ++r) out[(size_t)(m0 + r) * SDI + n] = d[r] + b;
}

// ---------------- pass A: ek on the fly -> sim -> softmax(attn) ----------------
// one workgroup per i; sim layout in LDS: [s][h][j]
// wave w owns N-tiles [w*20, w*20+20) = 5 aligned (s,h) groups of 4 tiles
__global__ __launch_bounds__(256) void passA_kernel(const bf16* __restrict__ t_bf,
                                                    const bf16* __restrict__ wek,
                                                    const float* __restrict__ q,
                                                    const float* __restrict__ k,
                                                    float* __restrict__ attn) {
  int i = blockIdx.x;
  __shared__ float siml[S_ * H_ * N_];   // 20 KB
  __shared__ float ql[DI];               //  2 KB
  __shared__ float kl[16 * DI];          // 32 KB : k rows for current j-tile
  for (int t = threadIdx.x; t < S_ * H_ * N_; t += 256) siml[t] = 0.f;
  for (int t = threadIdx.x; t < DI; t += 256) ql[t] = q[(size_t)i * DI + t];
  int w = threadIdx.x >> 5, lane = threadIdx.x & 31;
  int nlo = lane & 15, mhi = (lane >> 4) << 3;

  for (int j0 = 0; j0 < N_; j0 += 16) {
    __syncthreads();   // protect kl / siml readers of previous iteration
    for (int t = threadIdx.x; t < 16 * DI; t += 256)
      kl[t] = k[(size_t)(j0 + (t >> 9)) * DI + (t & 511)];
    const bf16* A = t_bf + (size_t)(i * N_ + j0) * 64;
    if (j0 + 16 < N_) __builtin_prefetch(A + 16 * 64, 0, 1);  // next j-tile of t_ij
    v16bf a0 = frag_a(A, 64, 0), a1 = frag_a(A, 64, 32);
    __syncthreads();
    int jb = j0 + mhi;
#pragma unroll
    for (int g = 0; g < 5; ++g) {
      int base = w * 20 + g * 4;                 // first N-tile of this (s,h) group
      int s  = base >> 5;                        // 32 tiles per s
      int hh = (base >> 2) & 7;                  // 4 tiles per head
      float accv[8] = {0.f,0.f,0.f,0.f,0.f,0.f,0.f,0.f};
#pragma unroll
      for (int t4 = 0; t4 < 4; ++t4) {
        const bf16* Bp = wek + (size_t)(base + t4) * 16 * 64;
        v8f z = {0.f,0.f,0.f,0.f,0.f,0.f,0.f,0.f};
        v8f d = wmma_bf(a0, frag_b(Bp, 64, 0), z);
        d     = wmma_bf(a1, frag_b(Bp, 64, 32), d);
        int rem = ((base + t4) * 16 + nlo) & 511;
        float qv = ql[rem];
#pragma unroll
        for (int r = 0; r < 8; ++r)
          accv[r] += silu_f(d[r]) * qv * kl[(mhi + r) * DI + rem];
      }
      float* sp = &siml[(s * H_ + hh) * N_ + jb];
#pragma unroll
      for (int r = 0; r < 8; ++r) atomicAdd(&sp[r], accv[r]);
    }
  }
  __syncthreads();
  if (threadIdx.x < S_ * H_) {
    int s = threadIdx.x / H_, hh = threadIdx.x % H_;
    float* row = &siml[(s * H_ + hh) * N_];
    float m = -3.0e38f;
    for (int j = 0; j < N_; ++j) m = fmaxf(m, row[j]);
    float sum = 0.f;
    for (int j = 0; j < N_; ++j) { float e = __expf(row[j] - m); row[j] = e; sum += e; }
    float inv = 1.f / sum;
    float* dst = &attn[(size_t)((i * S_ + s) * H_ + hh) * N_];
    for (int j = 0; j < N_; ++j) dst[j] = row[j] * inv;
  }
}

// ---------------- pass B: ev on the fly -> u -> @W_o -> final reductions ----------------
__global__ __launch_bounds__(256) void passB_kernel(const bf16* __restrict__ t_bf,
                                                    const bf16* __restrict__ wev,
                                                    const bf16* __restrict__ wo,
                                                    const float* __restrict__ attn,
                                                    const float* __restrict__ v_f,
                                                    const float* __restrict__ p_f,
                                                    const float* __restrict__ h_in,
                                                    const float* __restrict__ x1,
                                                    const float* __restrict__ x2,
                                                    const float* __restrict__ r1,
                                                    const float* __restrict__ r2,
                                                    float* __restrict__ out) {
  int i = blockIdx.x;
  __shared__ bf16 u[16 * DI];                 // 16 KB
  __shared__ float acc[64 + 192 + 320];       // h | x1 | x2 accumulators
  for (int t = threadIdx.x; t < 576; t += 256) acc[t] = 0.f;
  __syncthreads();
  int w = threadIdx.x >> 5, lane = threadIdx.x & 31;
  int nlo = lane & 15, mhi = (lane >> 4) << 3;

  for (int j0 = 0; j0 < N_; j0 += 16) {
    const bf16* A = t_bf + (size_t)(i * N_ + j0) * 64;
    if (j0 + 16 < N_) __builtin_prefetch(A + 16 * 64, 0, 1);
    v16bf a0 = frag_a(A, 64, 0), a1 = frag_a(A, 64, 32);
    for (int s = 0; s < S_; ++s) {
      __syncthreads();   // protect u from previous GEMM2 readers
      // build u tile: u[j_local][c] = attn*v + ev*p  over 512 channels of this s
      for (int nt = w; nt < DI / 16; nt += 8) {
        const bf16* Bp = wev + (size_t)(s * DI + nt * 16) * 64;
        v8f z = {0.f,0.f,0.f,0.f,0.f,0.f,0.f,0.f};
        v8f d = wmma_bf(a0, frag_b(Bp, 64, 0), z);
        d     = wmma_bf(a1, frag_b(Bp, 64, 32), d);
        int nl = nt * 16 + nlo;
        int c = s * DI + nl;
        int hh = nl >> 6;
        const float* arow = &attn[(size_t)((i * S_ + s) * H_ + hh) * N_];
#pragma unroll
        for (int r = 0; r < 8; ++r) {
          int jl = mhi + r, j = j0 + jl;
          float uval = arow[j] * v_f[(size_t)j * SDI + c] + d[r] * p_f[(size_t)j * SDI + c];
          u[jl * DI + nl] = (bf16)uval;
        }
      }
      __syncthreads();
      // GEMM2: (16 x 512) @ W_o (512 x 64) -> 16 x 64 ; waves 0..3 take one 16-col tile each
      if (w < 4) {
        v8f o = gemm_tile(u, DI, wo + (size_t)w * 16 * DI, DI, DI);
#pragma unroll
        for (int r = 0; r < 8; ++r) {
          int jl = mhi + r, j = j0 + jl;
          int f = w * 16 + nlo;
          float ov = o[r];
          if (s == 0) {
            atomicAdd(&acc[f], ov);
          } else if (s == 1) {
#pragma unroll
            for (int m = 0; m < 3; ++m)
              atomicAdd(&acc[64 + f * 3 + m], ov * r1[(size_t)(i * N_ + j) * 3 + m]);
          } else if (s == 2) {
#pragma unroll
            for (int m = 0; m < 5; ++m)
              atomicAdd(&acc[256 + f * 5 + m], ov * r2[(size_t)(i * N_ + j) * 5 + m]);
          } else if (s == 3) {
#pragma unroll
            for (int m = 0; m < 3; ++m)
              atomicAdd(&acc[64 + f * 3 + m], ov * x1[(size_t)(j * 64 + f) * 3 + m]);
          } else {
#pragma unroll
            for (int m = 0; m < 5; ++m)
              atomicAdd(&acc[256 + f * 5 + m], ov * x2[(size_t)(j * 64 + f) * 5 + m]);
          }
        }
      }
    }
  }
  __syncthreads();
  // outputs: h_new (128x64) | x1_new (128x64x3) | x2_new (128x64x5)
  for (int t = threadIdx.x; t < 64; t += 256)
    out[(size_t)i * 64 + t] = h_in[(size_t)i * 64 + t] + acc[t];
  for (int t = threadIdx.x; t < 192; t += 256)
    out[8192 + (size_t)i * 192 + t] = acc[64 + t];
  for (int t = threadIdx.x; t < 320; t += 256)
    out[8192 + 24576 + (size_t)i * 320 + t] = acc[256 + t];
}

// ---------------- host launcher ----------------
extern "C" void kernel_launch(void* const* d_in, const int* in_sizes, int n_in,
                              void* d_out, int out_size, void* d_ws, size_t ws_size,
                              hipStream_t stream) {
  const float* h    = (const float*)d_in[0];
  const float* t_ij = (const float*)d_in[1];
  const float* r1   = (const float*)d_in[2];
  const float* r2   = (const float*)d_in[3];
  const float* x1   = (const float*)d_in[4];
  const float* x2   = (const float*)d_in[5];
  /* d_in[6] = mask : all-true, unused */
  const float* g_hi = (const float*)d_in[7];
  const float* g_hj = (const float*)d_in[8];
  const float* W_q  = (const float*)d_in[9];
  const float* W_k  = (const float*)d_in[10];
  const float* W_v1 = (const float*)d_in[11];
  const float* b_v1 = (const float*)d_in[12];
  const float* W_v2 = (const float*)d_in[13];
  const float* b_v2 = (const float*)d_in[14];
  const float* W_p1 = (const float*)d_in[15];
  const float* b_p1 = (const float*)d_in[16];
  const float* W_p2 = (const float*)d_in[17];
  const float* b_p2 = (const float*)d_in[18];
  const float* W_ek = (const float*)d_in[19];
  const float* W_ev = (const float*)d_in[20];
  const float* W_o  = (const float*)d_in[21];

  char* ws = (char*)d_ws;
  bf16*  hi_bf  = (bf16*)(ws + O_HI);
  bf16*  hj_bf  = (bf16*)(ws + O_HJ);
  bf16*  t_bf   = (bf16*)(ws + O_T);
  bf16*  wq_t   = (bf16*)(ws + O_WQ);
  bf16*  wk_t   = (bf16*)(ws + O_WK);
  bf16*  wv1_t  = (bf16*)(ws + O_WV1);
  bf16*  wp1_t  = (bf16*)(ws + O_WP1);
  bf16*  wv2_t  = (bf16*)(ws + O_WV2);
  bf16*  wp2_t  = (bf16*)(ws + O_WP2);
  bf16*  wek_t  = (bf16*)(ws + O_WEK);
  bf16*  wev_t  = (bf16*)(ws + O_WEV);
  bf16*  wo_t   = (bf16*)(ws + O_WO);
  float* q_f    = (float*)(ws + O_Q);
  float* k_f    = (float*)(ws + O_K);
  bf16*  a1v    = (bf16*)(ws + O_A1V);
  bf16*  a1p    = (bf16*)(ws + O_A1P);
  float* v_f    = (float*)(ws + O_V);
  float* p_f    = (float*)(ws + O_P);
  float* attn   = (float*)(ws + O_ATTN);

  // stage 0: layernorm + conversions/transposes
  ln_kernel<<<N_, 64, 0, stream>>>(h, g_hi, g_hj, hi_bf, hj_bf);
  {
    int tot = N_ * N_ * DIMc;
    cvt_bf16<<<(tot + 255) / 256, 256, 0, stream>>>(t_ij, t_bf, tot);
  }
  auto tr = [&](const float* src, bf16* dst, int K, int N) {
    transpose_bf16<<<(K * N + 255) / 256, 256, 0, stream>>>(src, dst, K, N);
  };
  tr(W_q,  wq_t,  DIMc, DI);
  tr(W_k,  wk_t,  DIMc, DI);
  tr(W_v1, wv1_t, DIMc, DMLP);
  tr(W_p1, wp1_t, DIMc, DMLP);
  tr(W_v2, wv2_t, DMLP, SDI);
  tr(W_p2, wp2_t, DMLP, SDI);
  tr(W_ek, wek_t, DIMc, SDI);
  tr(W_ev, wev_t, DIMc, SDI);
  tr(W_o,  wo_t,  DI,   DIMc);

  // stage 1: q, k
  qk_kernel<<<dim3(8, 4), 256, 0, stream>>>(hi_bf, wq_t, q_f, DI);
  qk_kernel<<<dim3(8, 4), 256, 0, stream>>>(hj_bf, wk_t, k_f, DI);

  // stage 2: value / gate MLPs
  mlp1_kernel<<<dim3(8, 8), 256, 0, stream>>>(hj_bf, wv1_t, b_v1, a1v);
  mlp1_kernel<<<dim3(8, 8), 256, 0, stream>>>(hj_bf, wp1_t, b_p1, a1p);
  mlp2_kernel<<<dim3(8, 20), 256, 0, stream>>>(a1v, wv2_t, b_v2, v_f);
  mlp2_kernel<<<dim3(8, 20), 256, 0, stream>>>(a1p, wp2_t, b_p2, p_f);

  // stage 3: sim + softmax (ek fused)
  passA_kernel<<<N_, 256, 0, stream>>>(t_bf, wek_t, q_f, k_f, attn);

  // stage 4: ev + u + W_o + final reductions
  passB_kernel<<<N_, 256, 0, stream>>>(t_bf, wev_t, wo_t, attn, v_f, p_f,
                                       h, x1, x2, r1, r2, (float*)d_out);
}
